// CNNBCHDecoder_12893491822747
// MI455X (gfx1250) — compile-verified, matrix operand order checked by reading, not run on hardware
//
#include <hip/hip_runtime.h>

// ---- CDNA5 (gfx1250) vector types ----------------------------------------
typedef __attribute__((ext_vector_type(16))) _Float16 v16h;
typedef __attribute__((ext_vector_type(8)))  float    v8f;
typedef __attribute__((ext_vector_type(4)))  float    v4f;
typedef __attribute__((ext_vector_type(2)))  float    v2f;

#define MSG_LEN  1000   // groups per row (REP = 2)
#define CW_LEN   2000   // received bits per row
#define CHUNKS   500    // float4 chunks per row; each chunk = 2 groups
#define WAVES_PER_BLOCK 8

// One wave32 per row. Streaming, memory-bound:
//   - NT float4 loads  (global_load_b128, TH=NT)  : 16 B / lane, coalesced
//   - NT float2 stores (global_store_b64, TH=NT)  : 8 B / lane, coalesced
// Epilogue: 32-lane error reduction via a single v_wmma_f32_16x16x32_f16.
__global__ __launch_bounds__(256) void CNNBCHDecoder_gfx1250_kernel(
    const float* __restrict__ in,   // [batch, 2000]
    float* __restrict__ dec,        // [batch, 1000]
    float* __restrict__ errc,       // [batch]
    int batch)
{
  const int lane = threadIdx.x & 31;
  const int wave = threadIdx.x >> 5;
  const int row  = blockIdx.x * WAVES_PER_BLOCK + wave;
  if (row >= batch) return;  // wave-uniform: EXEC stays all-1 for live waves

  const float* rin  = in  + (size_t)row * CW_LEN;
  float*       rdec = dec + (size_t)row * MSG_LEN;

  float err = 0.0f;

  // 500 chunks, 16 strided iterations of 32 lanes (512 slots, tail masked).
  #pragma unroll
  for (int i = 0; i < 16; ++i) {
    const int c = i * 32 + lane;          // chunk index (2 groups per chunk)
    if (c < CHUNKS) {
      v4f v = __builtin_nontemporal_load((const v4f*)(rin + 4 * c));
      const bool b0 = v.x > 0.5f;
      const bool b1 = v.y > 0.5f;
      const bool b2 = v.z > 0.5f;
      const bool b3 = v.w > 0.5f;
      v2f d;
      d.x = (b0 && b1) ? 1.0f : 0.0f;     // majority (ones > 1) for REP=2
      d.y = (b2 && b3) ? 1.0f : 0.0f;
      err += (float)(b0 != b1) + (float)(b2 != b3);  // min(ones, 2-ones)
      __builtin_nontemporal_store(d, (v2f*)(rdec + 2 * c));
    }
  }

  // ---- Wave-wide sum of per-lane `err` via WMMA --------------------------
  // A (16x32 f16): every lane replicates its partial into its 16 slots, so
  // row m of A holds x_m in 16 K-slots and x_{m+16} in the other 16.
  // B (32x16 f16): uniform 1/16 (layout-invariant fill) cancels the 16x:
  //   D[m][n] = x_m + x_{m+16}   (exact: partials are integers <= 32).
  v16h a, b;
  const _Float16 e = (_Float16)err;
  #pragma unroll
  for (int k = 0; k < 16; ++k) { a[k] = e; b[k] = (_Float16)0.0625f; }
  v8f cacc = {};
  cacc = __builtin_amdgcn_wmma_f32_16x16x32_f16(
      /*neg_a=*/false, a, /*neg_b=*/false, b,
      /*c_mod=*/(short)0, cacc, /*reuse_a=*/false, /*reuse_b=*/false);

  // Lane n (<16) holds D[0..7][n]; lane n+16 holds D[8..15][n].
  float partial = cacc[0] + cacc[1] + cacc[2] + cacc[3] +
                  cacc[4] + cacc[5] + cacc[6] + cacc[7];
  float total = partial + __shfl_xor(partial, 16, 32);

  if (lane == 0) errc[row] = total;
}

extern "C" void kernel_launch(void* const* d_in, const int* in_sizes, int n_in,
                              void* d_out, int out_size, void* d_ws, size_t ws_size,
                              hipStream_t stream) {
  (void)n_in; (void)d_ws; (void)ws_size; (void)out_size;

  const float* in = (const float*)d_in[0];
  const int batch = in_sizes[0] / CW_LEN;            // 65536

  float* dec  = (float*)d_out;                       // [batch * 1000]
  float* errc = (float*)d_out + (size_t)batch * MSG_LEN;  // [batch]

  const int blocks = (batch + WAVES_PER_BLOCK - 1) / WAVES_PER_BLOCK;  // 8192
  CNNBCHDecoder_gfx1250_kernel<<<blocks, 32 * WAVES_PER_BLOCK, 0, stream>>>(
      in, dec, errc, batch);
}
